// ReAttention_57664230916959
// MI455X (gfx1250) — matrix operand order repaired and checked by reading, not compiled, not run on hardware
//
#include <hip/hip_runtime.h>
#include <hip/hip_bf16.h>

typedef __attribute__((ext_vector_type(16))) _Float16 v16h;
typedef __attribute__((ext_vector_type(8)))  _Float16 v8h;
typedef __attribute__((ext_vector_type(8)))  float    v8f;

#define B_  4
#define N_  1024
#define C_  768
#define H_  8
#define HD_ 96
#define SCALE_ 0.10206207261596575f   // 96^-0.5
#define EPS_   1e-3f
#define BPAD_  40                     // LDS column stride (halfs): 32 + 8 pad

// ---------------- WMMA fragment helpers (ISA 7.12.2 layouts, wave32) ----------------

// A-matrix 16x32 f16: lane m=lane%16; lanes<16 hold K {0..7,16..23}, lanes>=16 {8..15,24..31}
__device__ __forceinline__ v16h load_a_f16(const _Float16* __restrict__ p, int ld) {
  int lane = threadIdx.x & 31;
  int m  = lane & 15;
  int kb = (lane >> 4) << 3;                 // 0 or 8
  const _Float16* r = p + (size_t)m * ld + kb;
  v8h lo = *(const v8h*)r;                   // K = kb .. kb+7
  v8h hi = *(const v8h*)(r + 16);            // K = 16+kb .. 23+kb
  v16h a;
#pragma unroll
  for (int i = 0; i < 8; ++i) { a[i] = lo[i]; a[i + 8] = hi[i]; }
  return a;
}

// Same A layout but source is f32 (convert on load)
__device__ __forceinline__ v16h load_a_f32(const float* __restrict__ p, int ld) {
  int lane = threadIdx.x & 31;
  int m  = lane & 15;
  int kb = (lane >> 4) << 3;
  const float* r = p + (size_t)m * ld + kb;
  v16h a;
#pragma unroll
  for (int i = 0; i < 8; ++i) { a[i] = (_Float16)r[i]; a[i + 8] = (_Float16)r[16 + i]; }
  return a;
}

// B-matrix 32x16 f16 with B[k][n] = p[n*ld + k] (K contiguous per column).
// lane n=lane%16; lanes<16 hold K {0..7,16..23}, lanes>=16 hold K {8..15,24..31}.
__device__ __forceinline__ v16h load_b_kcont(const _Float16* __restrict__ p, int ld) {
  int lane = threadIdx.x & 31;
  int n  = lane & 15;
  int kb = (lane >> 4) << 3;                 // 0 or 8
  const _Float16* r = p + (size_t)n * ld + kb;
  v8h lo = *(const v8h*)r;
  v8h hi = *(const v8h*)(r + 16);
  v16h b;
#pragma unroll
  for (int i = 0; i < 8; ++i) { b[i] = lo[i]; b[i + 8] = hi[i]; }
  return b;
}

// Same B layout but from an LDS-staged tile with column stride BPAD_ halfs.
__device__ __forceinline__ v16h load_b_lds(const _Float16* p) {
  int lane = threadIdx.x & 31;
  int n  = lane & 15;
  int kb = (lane >> 4) << 3;
  const _Float16* r = p + n * BPAD_ + kb;
  v8h lo = *(const v8h*)r;
  v8h hi = *(const v8h*)(r + 16);
  v16h b;
#pragma unroll
  for (int i = 0; i < 8; ++i) { b[i] = lo[i]; b[i + 8] = hi[i]; }
  return b;
}

__device__ __forceinline__ v8f wmma16(v16h a, v16h b, v8f c) {
  return __builtin_amdgcn_wmma_f32_16x16x32_f16(false, a, false, b, (short)0, c,
                                                false, false);
}

__device__ __forceinline__ void zero8(v8f* acc, int n) {
  for (int t = 0; t < n; ++t)
#pragma unroll
    for (int i = 0; i < 8; ++i) acc[t][i] = 0.f;
}

// Async copy 32 bytes (16 halfs) global -> LDS, tracked by ASYNCcnt.
__device__ __forceinline__ void async_copy32(uint32_t lds_off, const _Float16* g) {
  asm volatile(
      "global_load_async_to_lds_b128 %0, %1, off\n\t"
      "global_load_async_to_lds_b128 %0, %1, off offset:16"
      :: "v"(lds_off), "v"(g) : "memory");
}

__device__ __forceinline__ void wait_async0() {
  asm volatile("s_wait_asynccnt 0x0" ::: "memory");
}

// ---------------- K1: per-patch 3x3 SAME conv -> q,k (row-major f16), v (transposed f16)
__global__ void __launch_bounds__(256) qkv_conv_kernel(
    const float* __restrict__ x,  const float* __restrict__ qw,
    const float* __restrict__ kw, const float* __restrict__ vw,
    _Float16* __restrict__ q16, _Float16* __restrict__ k16,
    _Float16* __restrict__ vT16) {
  __shared__ float sx[768];
  __shared__ float sw[3 * 81];
  int tid = threadIdx.x;
  int patch = blockIdx.x;                 // b*N + n
  int b = patch >> 10;
  int n = patch & (N_ - 1);
  const float* xp = x + (size_t)patch * C_;
#pragma unroll
  for (int j = 0; j < 3; ++j) sx[tid + j * 256] = xp[tid + j * 256];
  if (tid < 243) {
    float wv;
    if      (tid <  81) wv = qw[tid];
    else if (tid < 162) wv = kw[tid - 81];
    else                wv = vw[tid - 162];
    sw[tid] = wv;
  }
  __syncthreads();
  const float* swq = sw;
  const float* swk = sw + 81;
  const float* swv = sw + 162;
#pragma unroll
  for (int j = 0; j < 3; ++j) {
    int flat = tid + j * 256;            // (s0*16 + s1)*3 + co
    int co = flat % 3;
    int sp = flat / 3;
    int s1 = sp & 15;
    int s0 = sp >> 4;
    float aq = 0.f, ak = 0.f, av = 0.f;
#pragma unroll
    for (int dh = 0; dh < 3; ++dh) {
      int si = s0 + dh - 1;
      if (si < 0 || si >= 16) continue;
#pragma unroll
      for (int dw = 0; dw < 3; ++dw) {
        int sj = s1 + dw - 1;
        if (sj < 0 || sj >= 16) continue;
#pragma unroll
        for (int ci = 0; ci < 3; ++ci) {
          float xv = sx[(si * 16 + sj) * 3 + ci];
          int wi = ((dh * 3 + dw) * 3 + ci) * 3 + co;
          aq = fmaf(xv, swq[wi], aq);
          ak = fmaf(xv, swk[wi], ak);
          av = fmaf(xv, swv[wi], av);
        }
      }
    }
    int h  = flat / HD_;
    int hd = flat % HD_;
    int bh = b * H_ + h;
    q16[((size_t)bh * N_ + n) * HD_ + hd]  = (_Float16)aq;
    k16[((size_t)bh * N_ + n) * HD_ + hd]  = (_Float16)ak;
    vT16[((size_t)bh * HD_ + hd) * N_ + n] = (_Float16)av;
  }
}

// ---------------- K2: f32 -> f16 transpose convert (for rw, proj_w)
__global__ void __launch_bounds__(256) cvt_t_kernel(const float* __restrict__ src,
                                                    _Float16* __restrict__ dst,
                                                    int rows, int cols) {
  int idx = blockIdx.x * 256 + threadIdx.x;
  if (idx >= rows * cols) return;
  int r = idx / cols;
  int c = idx - r * cols;
  dst[(size_t)c * rows + r] = (_Float16)src[idx];
}

// ---------------- K3: attn logits = scale * q @ k^T  (per bh: 1024x1024, K=96)
__global__ void __launch_bounds__(256) qk_kernel(const _Float16* __restrict__ q16,
                                                 const _Float16* __restrict__ k16,
                                                 float* __restrict__ attn) {
  int bh = blockIdx.z;
  int w = threadIdx.x >> 5, lane = threadIdx.x & 31;
  int row0 = blockIdx.y * 128 + w * 16;
  int col0 = blockIdx.x * 128;
  const _Float16* qp = q16 + (size_t)bh * N_ * HD_ + (size_t)row0 * HD_;
  const _Float16* kp = k16 + (size_t)bh * N_ * HD_ + (size_t)col0 * HD_;
  v8f acc[8]; zero8(acc, 8);
  for (int kk = 0; kk < HD_; kk += 32) {
    v16h a = load_a_f16(qp + kk, HD_);
#pragma unroll
    for (int t = 0; t < 8; ++t) {
      v16h bm = load_b_kcont(kp + (size_t)(t * 16) * HD_ + kk, HD_);
      acc[t] = wmma16(a, bm, acc[t]);
    }
  }
  float* op = attn + (size_t)bh * N_ * N_;
  int rbase = row0 + ((lane >> 4) << 3);
  int nn = lane & 15;
#pragma unroll
  for (int t = 0; t < 8; ++t)
#pragma unroll
    for (int r = 0; r < 8; ++r)
      op[(size_t)(rbase + r) * N_ + col0 + t * 16 + nn] = acc[t][r] * SCALE_;
}

// ---------------- K4: row softmax (1024 wide) -> f16
__global__ void __launch_bounds__(256) softmax_kernel(const float* __restrict__ attn,
                                                      _Float16* __restrict__ sm16) {
  __shared__ float redm[8];
  __shared__ float reds[8];
  int row = blockIdx.x;
  int tid = threadIdx.x;
  const float* rp = attn + (size_t)row * N_;
  float vals[4];
#pragma unroll
  for (int j = 0; j < 4; ++j) vals[j] = rp[tid + j * 256];
  float m = fmaxf(fmaxf(vals[0], vals[1]), fmaxf(vals[2], vals[3]));
#pragma unroll
  for (int off = 16; off > 0; off >>= 1) m = fmaxf(m, __shfl_xor(m, off, 32));
  if ((tid & 31) == 0) redm[tid >> 5] = m;
  __syncthreads();
  float gm = redm[0];
#pragma unroll
  for (int i = 1; i < 8; ++i) gm = fmaxf(gm, redm[i]);
  float s = 0.f;
#pragma unroll
  for (int j = 0; j < 4; ++j) { vals[j] = __expf(vals[j] - gm); s += vals[j]; }
#pragma unroll
  for (int off = 16; off > 0; off >>= 1) s += __shfl_xor(s, off, 32);
  if ((tid & 31) == 0) reds[tid >> 5] = s;
  __syncthreads();
  float gs = 0.f;
#pragma unroll
  for (int i = 0; i < 8; ++i) gs += reds[i];
  float inv = 1.f / gs;
  _Float16* op = sm16 + (size_t)row * N_;
#pragma unroll
  for (int j = 0; j < 4; ++j) op[tid + j * 256] = (_Float16)(vals[j] * inv);
}

// ---------------- K5: re-attention GEMM (K=1024) + rb + BatchNorm -> final attn (f32)
// B panel staged in LDS via async-to-LDS copies, double buffered over the K loop.
__global__ void __launch_bounds__(256) reattn_kernel(
    const _Float16* __restrict__ sm16, const _Float16* __restrict__ rwT,
    const float* __restrict__ rb, const float* __restrict__ gamma,
    const float* __restrict__ beta, const float* __restrict__ mean,
    const float* __restrict__ var, float* __restrict__ attn_out) {
  __shared__ _Float16 sB[2][128 * BPAD_];        // [buf][col*BPAD_ + k]
  int bh = blockIdx.z;
  int tid = threadIdx.x;
  int w = tid >> 5, lane = tid & 31;
  int row0 = blockIdx.y * 128 + w * 16;
  int col0 = blockIdx.x * 128;
  const _Float16* ap = sm16 + (size_t)bh * N_ * N_ + (size_t)row0 * N_;

  // copy mapping: thread -> (column, 16-half chunk); 256 threads x 32B = 8KB slice
  int ccol  = tid >> 1;
  int chalf = (tid & 1) << 4;                    // 0 or 16 halfs
  const _Float16* gsrc = rwT + (size_t)(col0 + ccol) * N_ + chalf;
  uint32_t ldst = (uint32_t)(uintptr_t)&sB[0][ccol * BPAD_ + chalf];
  const uint32_t lstride = 128 * BPAD_ * sizeof(_Float16);

  v8f acc[8]; zero8(acc, 8);
  async_copy32(ldst, gsrc);                      // prologue: K-step 0 -> buf 0
  const int NSTEP = N_ / 32;
  for (int i = 0; i < NSTEP; ++i) {
    int cur = i & 1;
    wait_async0();                               // own copies for buf[cur] done
    __syncthreads();                             // everyone's copies done; prev reads done
    if (i + 1 < NSTEP)
      async_copy32(ldst + (cur ^ 1) * lstride, gsrc + (i + 1) * 32);
    v16h a = load_a_f16(ap + i * 32, N_);
    const _Float16* bbase = &sB[cur][0];
#pragma unroll
    for (int t = 0; t < 8; ++t) {
      v16h bm = load_b_lds(bbase + (t * 16) * BPAD_);
      acc[t] = wmma16(a, bm, acc[t]);
    }
  }
  float* op = attn_out + (size_t)bh * N_ * N_;
  int rbase = row0 + ((lane >> 4) << 3);
  int nn = lane & 15;
#pragma unroll
  for (int t = 0; t < 8; ++t) {
    int p = col0 + t * 16 + nn;
    float sc = gamma[p] * rsqrtf(var[p] + EPS_);
    float bi = beta[p] + (rb[p] - mean[p]) * sc;   // (acc+rb-mean)*sc+beta
#pragma unroll
    for (int r = 0; r < 8; ++r)
      op[(size_t)(rbase + r) * N_ + p] = acc[t][r] * sc + bi;
  }
}

// ---------------- K6: out_pre[b,n,h*HD+d] = attn(f32) @ v   (per bh: 1024x96, K=1024)
__global__ void __launch_bounds__(256) av_kernel(const float* __restrict__ attn,
                                                 const _Float16* __restrict__ vT16,
                                                 _Float16* __restrict__ out16) {
  int bh = blockIdx.y;
  int b = bh >> 3, h = bh & 7;
  int w = threadIdx.x >> 5, lane = threadIdx.x & 31;
  int row0 = blockIdx.x * 128 + w * 16;
  const float* ap = attn + (size_t)bh * N_ * N_ + (size_t)row0 * N_;
  const _Float16* vp = vT16 + (size_t)bh * HD_ * N_;
  v8f acc[6]; zero8(acc, 6);
  for (int kk = 0; kk < N_; kk += 32) {
    v16h a = load_a_f32(ap + kk, N_);
#pragma unroll
    for (int t = 0; t < 6; ++t) {
      v16h bm = load_b_kcont(vp + (size_t)(t * 16) * N_ + kk, N_);
      acc[t] = wmma16(a, bm, acc[t]);
    }
  }
  int rbase = row0 + ((lane >> 4) << 3);
  int nn = lane & 15;
#pragma unroll
  for (int t = 0; t < 6; ++t)
#pragma unroll
    for (int r = 0; r < 8; ++r)
      out16[((size_t)b * N_ + (rbase + r)) * C_ + h * HD_ + t * 16 + nn] =
          (_Float16)acc[t][r];
}

// ---------------- K7: out = out_pre @ proj_w + proj_b  (4096x768, K=768)
// Same async-LDS double-buffered B panel as K5.
__global__ void __launch_bounds__(256) proj_kernel(const _Float16* __restrict__ in16,
                                                   const _Float16* __restrict__ wT,
                                                   const float* __restrict__ bias,
                                                   float* __restrict__ out) {
  __shared__ _Float16 sB[2][128 * BPAD_];
  int tid = threadIdx.x;
  int w = tid >> 5, lane = tid & 31;
  int row0 = blockIdx.y * 128 + w * 16;
  int col0 = blockIdx.x * 128;
  const _Float16* ap = in16 + (size_t)row0 * C_;

  int ccol  = tid >> 1;
  int chalf = (tid & 1) << 4;
  const _Float16* gsrc = wT + (size_t)(col0 + ccol) * C_ + chalf;
  uint32_t ldst = (uint32_t)(uintptr_t)&sB[0][ccol * BPAD_ + chalf];
  const uint32_t lstride = 128 * BPAD_ * sizeof(_Float16);

  v8f acc[8]; zero8(acc, 8);
  async_copy32(ldst, gsrc);
  const int NSTEP = C_ / 32;
  for (int i = 0; i < NSTEP; ++i) {
    int cur = i & 1;
    wait_async0();
    __syncthreads();
    if (i + 1 < NSTEP)
      async_copy32(ldst + (cur ^ 1) * lstride, gsrc + (i + 1) * 32);
    v16h a = load_a_f16(ap + i * 32, C_);
    const _Float16* bbase = &sB[cur][0];
#pragma unroll
    for (int t = 0; t < 8; ++t) {
      v16h bm = load_b_lds(bbase + (t * 16) * BPAD_);
      acc[t] = wmma16(a, bm, acc[t]);
    }
  }
  int rbase = row0 + ((lane >> 4) << 3);
  int nn = lane & 15;
#pragma unroll
  for (int t = 0; t < 8; ++t) {
    int c = col0 + t * 16 + nn;
    float bv = bias[c];
#pragma unroll
    for (int r = 0; r < 8; ++r)
      out[(size_t)(rbase + r) * C_ + c] = acc[t][r] + bv;
  }
}

// ---------------------------------------------------------------------------

extern "C" void kernel_launch(void* const* d_in, const int* in_sizes, int n_in,
                              void* d_out, int out_size, void* d_ws, size_t ws_size,
                              hipStream_t stream) {
  (void)in_sizes; (void)n_in; (void)out_size; (void)ws_size;
  const float* x     = (const float*)d_in[0];
  const float* qw    = (const float*)d_in[1];
  const float* kw    = (const float*)d_in[2];
  const float* vw    = (const float*)d_in[3];
  const float* rw    = (const float*)d_in[4];
  const float* rb    = (const float*)d_in[5];
  const float* gamma = (const float*)d_in[6];
  const float* beta  = (const float*)d_in[7];
  const float* mean  = (const float*)d_in[8];
  const float* var   = (const float*)d_in[9];
  const float* pw    = (const float*)d_in[10];
  const float* pb    = (const float*)d_in[11];

  float* out      = (float*)d_out;                          // [B,N,C]
  float* attn_out = out + (size_t)B_ * N_ * C_;             // [B,H,N,N]

  char* ws = (char*)d_ws;
  size_t off = 0;
  const size_t qkv_sz  = (size_t)B_ * H_ * N_ * HD_ * sizeof(_Float16);
  _Float16* q16  = (_Float16*)(ws + off); off += qkv_sz;
  _Float16* k16  = (_Float16*)(ws + off); off += qkv_sz;
  _Float16* vT16 = (_Float16*)(ws + off); off += qkv_sz;
  _Float16* rwT  = (_Float16*)(ws + off); off += (size_t)N_ * N_ * sizeof(_Float16);
  _Float16* pwT  = (_Float16*)(ws + off); off += (size_t)C_ * C_ * sizeof(_Float16);
  _Float16* o16  = (_Float16*)(ws + off); off += (size_t)B_ * N_ * C_ * sizeof(_Float16);
  _Float16* sm16 = (_Float16*)(ws + off); off += (size_t)B_ * H_ * N_ * N_ * sizeof(_Float16);

  // K1: conv -> q,k,vT (f16)
  qkv_conv_kernel<<<B_ * N_, 256, 0, stream>>>(x, qw, kw, vw, q16, k16, vT16);
  // K2: weight transpose-converts
  cvt_t_kernel<<<(N_ * N_ + 255) / 256, 256, 0, stream>>>(rw, rwT, N_, N_);
  cvt_t_kernel<<<(C_ * C_ + 255) / 256, 256, 0, stream>>>(pw, pwT, C_, C_);
  // K3: logits (stage in d_out attn region)
  qk_kernel<<<dim3(8, 8, B_ * H_), 256, 0, stream>>>(q16, k16, attn_out);
  // K4: softmax -> f16 (ws)
  softmax_kernel<<<B_ * H_ * N_, 256, 0, stream>>>(attn_out, sm16);
  // K5: re-attention + BN -> final attn (d_out), async-LDS staged B
  reattn_kernel<<<dim3(8, 8, B_ * H_), 256, 0, stream>>>(sm16, rwT, rb, gamma, beta,
                                                         mean, var, attn_out);
  // K6: attn @ v -> pre-proj f16
  av_kernel<<<dim3(8, B_ * H_), 256, 0, stream>>>(attn_out, vT16, o16);
  // K7: projection -> d_out, async-LDS staged B
  proj_kernel<<<dim3(6, 32), 256, 0, stream>>>(o16, pwT, pb, out);
}
